// RNNCRFModel_23493471109681
// MI455X (gfx1250) — compile-verified, matrix operand order checked by reading, not run on hardware
//
#include <hip/hip_runtime.h>
#include <hip/hip_bf16.h>

// ---------------------------------------------------------------------------
// BiLSTM-CRF for MI455X (gfx1250), wave32 + WMMA f32_16x16x32_f16.
// B=64, T=512, E=256, H=512, K=15, V=50.
// ---------------------------------------------------------------------------

typedef __attribute__((ext_vector_type(16))) _Float16 v16h;
typedef __attribute__((ext_vector_type(8)))  float    v8f;

#define Bq   64
#define Tq   512
#define Eq   256
#define Hq   512
#define G4   2048      // 4*H
#define H2   1024      // 2*H
#define Kq   15
#define LDH  520       // padded LDS row stride (halves) for h state
#define LDA  40        // padded LDS row stride (halves) for staged GEMM A tile

static __device__ __forceinline__ v8f wmma16(v16h a, v16h b, v8f c) {
  return __builtin_amdgcn_wmma_f32_16x16x32_f16(
      /*neg_a=*/false, a, /*neg_b=*/false, b,
      /*c_mod=*/(short)0, c, /*reuse_a=*/false, /*reuse_b=*/false);
}

// Load a 16x32 f16 A/B-style fragment from a row-major [rows, ld] matrix.
// rc0 = first row (A) / first output column (B, since W is [N,K] row-major).
static __device__ __forceinline__ v16h load_frag_k16(
    const _Float16* __restrict__ base, size_t ld, int rc0, int k0) {
  const int l = threadIdx.x & 31;
  const _Float16* p = base + (size_t)(rc0 + (l & 15)) * ld + k0 + ((l & 16) ? 8 : 0);
  v16h r;
#pragma unroll
  for (int e = 0; e < 8; ++e) r[e] = p[e];
#pragma unroll
  for (int e = 0; e < 8; ++e) r[8 + e] = p[16 + e];
  return r;
}

static __device__ __forceinline__ float sigm(float x) {
  return 1.0f / (1.0f + __expf(-x));
}

// ---------------------------------------------------------------------------
// Prep kernels
// ---------------------------------------------------------------------------
__global__ void cvt_f32_to_f16_kernel(const float* __restrict__ s,
                                      _Float16* __restrict__ d, int n) {
  int i = blockIdx.x * 256 + threadIdx.x;
  if (i < n) d[i] = (_Float16)s[i];
}

__global__ void prep_misc_kernel(const float* __restrict__ linw,
                                 const float* __restrict__ linb,
                                 const float* __restrict__ bih0,
                                 const float* __restrict__ bhh0,
                                 const float* __restrict__ bih1,
                                 const float* __restrict__ bhh1,
                                 _Float16* __restrict__ linwf,
                                 float* __restrict__ linbf,
                                 float* __restrict__ bias0,
                                 float* __restrict__ bias1) {
  int i = blockIdx.x * 256 + threadIdx.x;
  if (i < 16 * H2) {                          // pad lin_w [15,1024] -> [16,1024]
    int r = i / H2, c = i % H2;
    linwf[i] = (_Float16)((r < Kq) ? linw[r * H2 + c] : 0.0f);
  }
  if (i < 16) linbf[i] = (i < Kq) ? linb[i] : 0.0f;
  if (i < 2 * G4) {
    bias0[i] = bih0[i] + bhh0[i];
    bias1[i] = bih1[i] + bhh1[i];
  }
}

// Embedding gather -> xe [T*B, E] f16  (row = t*B + b)
__global__ void embed_kernel(const int* __restrict__ x,
                             const float* __restrict__ emb,
                             _Float16* __restrict__ xe) {
  size_t i = (size_t)blockIdx.x * 256 + threadIdx.x;   // over T*B*E
  int e  = (int)(i & (Eq - 1));
  int tb = (int)(i >> 8);
  int t = tb >> 6, b = tb & 63;
  int tok = x[b * Tq + t];
  xe[i] = (_Float16)emb[(size_t)tok * Eq + e];
}

// ---------------------------------------------------------------------------
// Input-projection GEMM with async-staged A tiles:
//   out[dir] = A[M][K] x W[dir][2048][K]^T + bias, stored in WMMA C-fragment
//   major layout [t][gate][j][m][lane][8] f16 so the LSTM C-init is one
//   contiguous 16B load per gate tile.
// Block: 256 threads (8 waves, 2x4), block tile 64x128, wave tile 32x32.
// A k-slices (64x32 f16 = 4KB) staged in LDS via GLOBAL_LOAD_ASYNC_TO_LDS,
// double buffered, one tile in flight overlapping WMMA.
// ---------------------------------------------------------------------------
__launch_bounds__(256)
__global__ void gemm_xw_kernel(const _Float16* __restrict__ A,
                               const _Float16* __restrict__ W2,   // [2][2048][K]
                               const float* __restrict__ bias2,   // [2][2048]
                               _Float16* __restrict__ out2,       // [2][...] frag-major
                               int M, int K) {
  __shared__ _Float16 sA[2][64 * LDA];     // 2 x 5 KB, padded rows

  const int dir = blockIdx.z;
  const _Float16* W = W2 + (size_t)dir * G4 * K;
  const float* bd = bias2 + dir * G4;
  _Float16* od = out2 + (size_t)dir * M * G4;

  const int wave = threadIdx.x >> 5;
  const int lane = threadIdx.x & 31;
  const int wm = wave & 1, wn = wave >> 1;
  const int row0 = blockIdx.x * 64 + wm * 32;    // global M base of wave tile
  const int col0 = blockIdx.y * 128 + wn * 32;   // N base of wave tile
  const int nk = K / 32;

  // async A-tile copy mapping: 256 threads x 16B = 4KB (64 rows x 32 halves)
  const int row_cp = threadIdx.x >> 2;           // 0..63
  const int quad   = threadIdx.x & 3;            // 0..3 (16B chunks)
  const _Float16* gsrc =
      A + (size_t)(blockIdx.x * 64 + row_cp) * K + quad * 8;
  const unsigned ldsoff0 =
      (unsigned)(size_t)&sA[0][row_cp * LDA + quad * 8];
  const unsigned ldsoff1 =
      (unsigned)(size_t)&sA[1][row_cp * LDA + quad * 8];

  v8f acc[2][2];
#pragma unroll
  for (int a = 0; a < 2; ++a)
#pragma unroll
    for (int b = 0; b < 2; ++b)
#pragma unroll
      for (int r = 0; r < 8; ++r) acc[a][b][r] = 0.0f;

  // prime tile 0
  {
    const _Float16* src = gsrc;
    asm volatile("global_load_async_to_lds_b128 %0, %1, off"
                 :: "v"(ldsoff0), "v"(src) : "memory");
    asm volatile("s_wait_asynccnt 0x0" ::: "memory");
  }
  __syncthreads();

  for (int kb = 0; kb < nk; ++kb) {
    if (kb + 1 < nk) {                       // issue next tile (1 in flight)
      unsigned dst = ((kb + 1) & 1) ? ldsoff1 : ldsoff0;
      const _Float16* src = gsrc + (size_t)(kb + 1) * 32;
      asm volatile("global_load_async_to_lds_b128 %0, %1, off"
                   :: "v"(dst), "v"(src) : "memory");
    }
    const _Float16* Ab = sA[kb & 1];
    v16h a0 = load_frag_k16(Ab, LDA, wm * 32,      0);
    v16h a1 = load_frag_k16(Ab, LDA, wm * 32 + 16, 0);
    v16h b0 = load_frag_k16(W, K, col0,      kb * 32);
    v16h b1 = load_frag_k16(W, K, col0 + 16, kb * 32);
    acc[0][0] = wmma16(a0, b0, acc[0][0]);
    acc[0][1] = wmma16(a0, b1, acc[0][1]);
    acc[1][0] = wmma16(a1, b0, acc[1][0]);
    acc[1][1] = wmma16(a1, b1, acc[1][1]);
    asm volatile("s_wait_asynccnt 0x0" ::: "memory");
    __syncthreads();
  }

  // epilogue: bias add + store in fragment-major layout
  const int n = lane & 15;
#pragma unroll
  for (int mi = 0; mi < 2; ++mi)
#pragma unroll
    for (int ni = 0; ni < 2; ++ni) {
      const int R = (row0 + 16 * mi) >> 4;       // global M tile index
      const int tt = R >> 2, m = R & 3;          // t, batch-tile
      const int C = (col0 + 16 * ni) >> 4;       // N tile index
      const int gate = C >> 5, j = C & 31;
      const float bv = bd[col0 + 16 * ni + n];
      _Float16* tp =
          od + ((((size_t)tt * 128 + gate * 32 + j) * 4 + m) * 32 + lane) * 8;
#pragma unroll
      for (int r = 0; r < 8; ++r) tp[r] = (_Float16)(acc[mi][ni][r] + bv);
    }
}

// ---------------------------------------------------------------------------
// Persistent BiLSTM layer: grid.x = direction (2 blocks), 512 threads each.
// Whole T-loop inside the kernel. h double-buffered in LDS; c in registers.
// xw is fragment-major so gate-accumulator init is one b128 load per tile.
// Next timestep's xw slab is prefetched (global_prefetch_b8) during compute.
// ---------------------------------------------------------------------------
__launch_bounds__(512, 1)
__global__ void lstm_layer_kernel(const _Float16* __restrict__ xw2,  // [2][T][frag-major]
                                  const _Float16* __restrict__ whh2, // [2][2048][512]
                                  _Float16* __restrict__ hs) {       // [T*B][1024]
  __shared__ _Float16 sh_h[2][Bq * LDH];   // 2 x 66.5 KB

  const int dir = blockIdx.x;
  const _Float16* xw = xw2 + (size_t)dir * Tq * Bq * G4;
  const _Float16* W  = whh2 + (size_t)dir * G4 * Hq;

  const int tid  = threadIdx.x;
  const int wave = tid >> 5;
  const int lane = tid & 31;
  const int mb   = (tid & 16) ? 8 : 0;
  const int n    = tid & 15;

  for (int i = tid; i < Bq * LDH; i += 512) sh_h[0][i] = (_Float16)0.0f;
  __syncthreads();

  v8f c_t[8];                               // persistent cell state tiles
#pragma unroll
  for (int i = 0; i < 8; ++i)
#pragma unroll
    for (int r = 0; r < 8; ++r) c_t[i][r] = 0.0f;

  int p = 0;
  for (int s = 0; s < Tq; ++s) {
    const int t = dir ? (Tq - 1 - s) : s;
    const _Float16* xwt = xw + (size_t)t * Bq * G4;   // frag-major slab, 256KB

    // prefetch next timestep's xw slab while this step computes
    if (s + 1 < Tq) {
      const int tn = dir ? (Tq - 2 - s) : (s + 1);
      const char* pf = (const char*)(xw + (size_t)tn * Bq * G4) + tid * 512;
#pragma unroll
      for (int q = 0; q < 4; ++q) __builtin_prefetch(pf + q * 128, 0, 3);
    }

#pragma unroll 1
    for (int i = 0; i < 8; ++i) {
      const int g = wave * 8 + i;           // 0..127
      const int m = g >> 5;                 // batch tile 0..3
      const int j = g & 31;                 // hidden tile 0..31
      const int row0 = m * 16;
      const int colh = j * 16;

      v8f acc[4];
#pragma unroll
      for (int gate = 0; gate < 4; ++gate) {    // C-init: one 16B load/tile
        const _Float16* cb =
            xwt + (((size_t)gate * 32 + j) * 4 + m) * 32 * 8 + lane * 8;
#pragma unroll
        for (int r = 0; r < 8; ++r) acc[gate][r] = (float)cb[r];
      }

#pragma unroll 2
      for (int kb = 0; kb < Hq / 32; ++kb) {
        v16h a = load_frag_k16(sh_h[p], LDH, row0, kb * 32);
#pragma unroll
        for (int gate = 0; gate < 4; ++gate) {
          v16h b = load_frag_k16(W, Hq, gate * Hq + colh, kb * 32);
          acc[gate] = wmma16(a, b, acc[gate]);
        }
      }

#pragma unroll
      for (int r = 0; r < 8; ++r) {         // in-register gate nonlinearity
        float iv = acc[0][r], fv = acc[1][r], gv = acc[2][r], ov = acc[3][r];
        float c = c_t[i][r];
        c = sigm(fv) * c + sigm(iv) * tanhf(gv);
        float h = sigm(ov) * tanhf(c);
        c_t[i][r] = c;
        int row = row0 + mb + r;            // batch index
        int col = colh + n;                 // hidden index
        _Float16 hh = (_Float16)h;
        sh_h[p ^ 1][row * LDH + col] = hh;
        hs[((size_t)t * Bq + row) * H2 + dir * Hq + col] = hh;
      }
    }
    __syncthreads();
    p ^= 1;
  }
}

// ---------------------------------------------------------------------------
// Emissions: em[M][16] f32 = hs1[M][1024] x linwf[16][1024]^T + linb
// ---------------------------------------------------------------------------
__launch_bounds__(256)
__global__ void gemm_em_kernel(const _Float16* __restrict__ A,
                               const _Float16* __restrict__ Wl,
                               const float* __restrict__ bias,
                               float* __restrict__ em, int M) {
  const int wave = threadIdx.x >> 5;
  const int row0 = blockIdx.x * 128 + wave * 16;
  v8f acc;
#pragma unroll
  for (int r = 0; r < 8; ++r) acc[r] = 0.0f;
  for (int kb = 0; kb < H2 / 32; ++kb) {
    v16h a = load_frag_k16(A,  H2, row0, kb * 32);
    v16h b = load_frag_k16(Wl, H2, 0,    kb * 32);
    acc = wmma16(a, b, acc);
  }
  const int mb = (threadIdx.x & 16) ? 8 : 0;
  const int n  = threadIdx.x & 15;
  const float bv = bias[n];
#pragma unroll
  for (int r = 0; r < 8; ++r)
    em[(size_t)(row0 + mb + r) * 16 + n] = acc[r] + bv;
}

// ---------------------------------------------------------------------------
// CRF NLL: one block of 64 threads, thread b handles batch b.
// ---------------------------------------------------------------------------
__launch_bounds__(64)
__global__ void crf_kernel(const float* __restrict__ em,    // [T*B][16]
                           const int* __restrict__ tags,    // [B][T]
                           const float* __restrict__ cs,
                           const float* __restrict__ ce,
                           const float* __restrict__ ct,    // [15][15]
                           float* __restrict__ out) {
  __shared__ float st[Kq * Kq];
  __shared__ float ssum[Bq];
  const int b = threadIdx.x;
  for (int i = threadIdx.x; i < Kq * Kq; i += 64) st[i] = ct[i];
  __syncthreads();

  const int* tg = tags + b * Tq;

  // numerator
  int pv = tg[0];
  float num = cs[pv] + em[(size_t)b * 16 + pv];
  for (int t = 1; t < Tq; ++t) {
    int cur = tg[t];
    num += em[((size_t)t * Bq + b) * 16 + cur] + st[pv * Kq + cur];
    pv = cur;
  }
  num += ce[tg[Tq - 1]];

  // denominator (forward algorithm)
  float sc[Kq];
#pragma unroll
  for (int k = 0; k < Kq; ++k) sc[k] = cs[k] + em[(size_t)b * 16 + k];
  for (int t = 1; t < Tq; ++t) {
    const float* et = em + ((size_t)t * Bq + b) * 16;
    float ns[Kq];
#pragma unroll
    for (int k2 = 0; k2 < Kq; ++k2) {
      float mx = -3.4e38f;
#pragma unroll
      for (int k1 = 0; k1 < Kq; ++k1) mx = fmaxf(mx, sc[k1] + st[k1 * Kq + k2]);
      float sum = 0.0f;
#pragma unroll
      for (int k1 = 0; k1 < Kq; ++k1) sum += __expf(sc[k1] + st[k1 * Kq + k2] - mx);
      ns[k2] = mx + __logf(sum) + et[k2];
    }
#pragma unroll
    for (int k = 0; k < Kq; ++k) sc[k] = ns[k];
  }
  float mx = -3.4e38f;
#pragma unroll
  for (int k = 0; k < Kq; ++k) mx = fmaxf(mx, sc[k] + ce[k]);
  float sum = 0.0f;
#pragma unroll
  for (int k = 0; k < Kq; ++k) sum += __expf(sc[k] + ce[k] - mx);
  float den = mx + __logf(sum);

  ssum[b] = num - den;
  __syncthreads();
  if (b == 0) {
    float tot = 0.0f;
    for (int i = 0; i < Bq; ++i) tot += ssum[i];
    out[0] = -tot;
  }
}

// ---------------------------------------------------------------------------
// Host side
// ---------------------------------------------------------------------------
extern "C" void kernel_launch(void* const* d_in, const int* in_sizes, int n_in,
                              void* d_out, int out_size, void* d_ws, size_t ws_size,
                              hipStream_t stream) {
  const int*   x        = (const int*)  d_in[0];
  const int*   tags     = (const int*)  d_in[1];
  const float* emb      = (const float*)d_in[2];
  const float* w_ih_l0  = (const float*)d_in[3];
  const float* w_hh_l0  = (const float*)d_in[4];
  const float* b_ih_l0  = (const float*)d_in[5];
  const float* b_hh_l0  = (const float*)d_in[6];
  const float* w_ih_l1  = (const float*)d_in[7];
  const float* w_hh_l1  = (const float*)d_in[8];
  const float* b_ih_l1  = (const float*)d_in[9];
  const float* b_hh_l1  = (const float*)d_in[10];
  const float* lin_w    = (const float*)d_in[11];
  const float* lin_b    = (const float*)d_in[12];
  const float* crf_s    = (const float*)d_in[13];
  const float* crf_e    = (const float*)d_in[14];
  const float* crf_t    = (const float*)d_in[15];
  float* out = (float*)d_out;

  const size_t TB = (size_t)Tq * Bq;     // 32768
  char* ws = (char*)d_ws;
  size_t o = 0;
  auto take = [&](size_t bytes) { char* p = ws + o; o = (o + bytes + 255) & ~(size_t)255; return p; };

  _Float16* xe     = (_Float16*)take(TB * Eq * 2);                 // 16 MB
  _Float16* wih0f  = (_Float16*)take((size_t)2 * G4 * Eq * 2);     // 2 MB
  _Float16* whh0f  = (_Float16*)take((size_t)2 * G4 * Hq * 2);     // 4 MB
  _Float16* wih1f  = (_Float16*)take((size_t)2 * G4 * H2 * 2);     // 8 MB
  _Float16* whh1f  = (_Float16*)take((size_t)2 * G4 * Hq * 2);     // 4 MB
  _Float16* linwf  = (_Float16*)take((size_t)16 * H2 * 2);
  float*    linbf  = (float*)   take(16 * 4);
  float*    bias0  = (float*)   take((size_t)2 * G4 * 4);
  float*    bias1  = (float*)   take((size_t)2 * G4 * 4);
  _Float16* xw     = (_Float16*)take((size_t)2 * TB * G4 * 2);     // 256 MB (reused)
  _Float16* hs0    = (_Float16*)take(TB * H2 * 2);                 // 64 MB
  _Float16* hs1    = (_Float16*)take(TB * H2 * 2);                 // 64 MB
  float*    em     = (float*)   take(TB * 16 * 4);                 // 2 MB
  (void)ws_size; (void)n_in; (void)in_sizes; (void)out_size;

  // weight conversion + bias prep
  {
    int n0 = 2 * G4 * Eq, n1 = 2 * G4 * Hq, n2 = 2 * G4 * H2;
    cvt_f32_to_f16_kernel<<<(n0 + 255) / 256, 256, 0, stream>>>(w_ih_l0, wih0f, n0);
    cvt_f32_to_f16_kernel<<<(n1 + 255) / 256, 256, 0, stream>>>(w_hh_l0, whh0f, n1);
    cvt_f32_to_f16_kernel<<<(n2 + 255) / 256, 256, 0, stream>>>(w_ih_l1, wih1f, n2);
    cvt_f32_to_f16_kernel<<<(n1 + 255) / 256, 256, 0, stream>>>(w_hh_l1, whh1f, n1);
    prep_misc_kernel<<<64, 256, 0, stream>>>(lin_w, lin_b, b_ih_l0, b_hh_l0,
                                             b_ih_l1, b_hh_l1, linwf, linbf,
                                             bias0, bias1);
  }

  // embedding -> [T*B, E] f16
  embed_kernel<<<(int)(TB * Eq / 256), 256, 0, stream>>>(x, emb, xe);

  // layer 0: input projection (K=256), then persistent recurrence
  gemm_xw_kernel<<<dim3(TB / 64, G4 / 128, 2), 256, 0, stream>>>(
      xe, wih0f, bias0, xw, (int)TB, Eq);
  lstm_layer_kernel<<<2, 512, 0, stream>>>(xw, whh0f, hs0);

  // layer 1: input projection (K=1024), then persistent recurrence
  gemm_xw_kernel<<<dim3(TB / 64, G4 / 128, 2), 256, 0, stream>>>(
      hs0, wih1f, bias1, xw, (int)TB, H2);
  lstm_layer_kernel<<<2, 512, 0, stream>>>(xw, whh1f, hs1);

  // emissions + CRF
  gemm_em_kernel<<<(int)(TB / 128), 256, 0, stream>>>(hs1, linwf, linbf, em, (int)TB);
  crf_kernel<<<1, 64, 0, stream>>>(em, tags, crf_s, crf_e, crf_t, out);
}
